// Attention_69630009803003
// MI455X (gfx1250) — matrix operand order repaired and checked by reading, not compiled, not run on hardware
//
#include <hip/hip_runtime.h>
#include <hip/hip_bf16.h>

typedef __bf16 bf16;
typedef __attribute__((ext_vector_type(16))) __bf16 v16bf;
typedef __attribute__((ext_vector_type(8)))  __bf16 v8bf;
typedef __attribute__((ext_vector_type(8)))  float  v8f;

union FragU { v16bf v; v8bf h[2]; };

__device__ __forceinline__ bf16 f2bf(float f) { return (bf16)f; }

__device__ __forceinline__ v8bf cvt8(float4 f0, float4 f1) {
    v8bf h;
    h[0] = (bf16)f0.x; h[1] = (bf16)f0.y; h[2] = (bf16)f0.z; h[3] = (bf16)f0.w;
    h[4] = (bf16)f1.x; h[5] = (bf16)f1.y; h[6] = (bf16)f1.z; h[7] = (bf16)f1.w;
    return h;
}

// ---------------------------------------------------------------------------
// B=64, N=197, C=768, H=12, hd=64.  NQ=208 (q pad), NK=224 (k/v pad)
// ---------------------------------------------------------------------------

// Zero padding rows of q (n in [197,208)) and k/v (n in [197,224)).
__global__ void zero_pads(bf16* qbuf, bf16* kbuf, bf16* vbuf) {
    int i = blockIdx.x * blockDim.x + threadIdx.x;
    const int KV = 768 * 27 * 64;
    if (i < KV) {
        int bh  = i / (27 * 64);
        int rem = i % (27 * 64);
        int n   = 197 + rem / 64;
        int d   = rem % 64;
        kbuf[((size_t)bh * 224 + n) * 64 + d] = f2bf(0.f);
        vbuf[((size_t)bh * 64 + d) * 224 + n] = f2bf(0.f);
    }
    const int QP = 768 * 11 * 64;
    if (i < QP) {
        int bh  = i / (11 * 64);
        int rem = i % (11 * 64);
        int n   = 197 + rem / 64;
        int d   = rem % 64;
        qbuf[((size_t)bh * 208 + n) * 64 + d] = f2bf(0.f);
    }
}

// Precompute padded relative-position bias: rpb[h][208][224], mask = -3e38.
__global__ void rpb_kernel(const float* __restrict__ rel_table,
                           const int* __restrict__ rel_index,
                           float* __restrict__ rpb) {
    int i = blockIdx.x * blockDim.x + threadIdx.x;
    const int TOT = 12 * 208 * 224;
    if (i >= TOT) return;
    int h   = i / (208 * 224);
    int rem = i % (208 * 224);
    int n   = rem / 224;
    int m   = rem % 224;
    float v = -3.0e38f;
    if (n < 197 && m < 197)
        v = rel_table[rel_index[n * 197 + m] * 12 + h];
    rpb[i] = v;
}

// ---------------------------------------------------------------------------
// QKV GEMM: [12608 x 768] f32 x [2304 x 768]^T f32 -> q/k/vT bf16 (scattered).
// Block = 64(M) x 128(N), 8 waves (4x2), each wave 16x64 = 4 accumulators.
// ---------------------------------------------------------------------------
__global__ __launch_bounds__(256)
void qkv_gemm(const float* __restrict__ x, const float* __restrict__ w,
              const float* __restrict__ qb, const float* __restrict__ vb,
              bf16* __restrict__ qbuf, bf16* __restrict__ kbuf, bf16* __restrict__ vbuf)
{
    __shared__ bf16 a_lds[64 * 32];
    __shared__ bf16 b_lds[128 * 32];
    const int t    = threadIdx.x;
    const int lane = t & 31;
    const int wave = t >> 5;
    const int m0   = (int)blockIdx.x * 64;
    const int n0   = (int)blockIdx.y * 128;
    const int wm   = wave >> 1;  // 0..3 : 16-row strip
    const int wn   = wave & 1;   // 0..1 : 64-col half

    v8f acc[4] = {};

    const int ra = t >> 2, ka = (t & 3) * 8;    // A: 8 f32 / thread
    const int rb = t >> 1, kb = (t & 1) * 16;   // B: 16 f32 / thread
    const int abase = (lane < 16) ? 0 : 16;
    const int bbase = (lane < 16) ? 0 : 16;
    const int arow  = wm * 16 + (lane & 15);

    for (int k0 = 0; k0 < 768; k0 += 32) {
        const float* sa = x + (size_t)(m0 + ra) * 768 + k0 + ka;
        const float* sb = w + (size_t)(n0 + rb) * 768 + k0 + kb;
        {   // A tile: x f32 -> bf16 LDS
            float4 f0 = *(const float4*)(sa);
            float4 f1 = *(const float4*)(sa + 4);
            *(v8bf*)(a_lds + ra * 32 + ka) = cvt8(f0, f1);
        }
        {   // B tile: qkv_w [outcol][K] -> [n][k] LDS
            float4 g0 = *(const float4*)(sb);
            float4 g1 = *(const float4*)(sb + 4);
            float4 g2 = *(const float4*)(sb + 8);
            float4 g3 = *(const float4*)(sb + 12);
            *(v8bf*)(b_lds + rb * 32 + kb)     = cvt8(g0, g1);
            *(v8bf*)(b_lds + rb * 32 + kb + 8) = cvt8(g2, g3);
        }
        if (k0 + 32 < 768) {   // prefetch next K tile -> global_prefetch_b8
            __builtin_prefetch(sa + 32, 0, 3);
            __builtin_prefetch(sb + 32, 0, 3);
        }
        __syncthreads();

        FragU af;
        af.h[0] = *(const v8bf*)(a_lds + arow * 32 + abase);
        af.h[1] = *(const v8bf*)(a_lds + arow * 32 + abase + 8);
        #pragma unroll
        for (int ti = 0; ti < 4; ++ti) {
            int bc = wn * 64 + ti * 16 + (lane & 15);
            FragU bfr;
            bfr.h[0] = *(const v8bf*)(b_lds + bc * 32 + bbase);
            bfr.h[1] = *(const v8bf*)(b_lds + bc * 32 + bbase + 8);
            acc[ti] = __builtin_amdgcn_wmma_f32_16x16x32_bf16(false, af.v, false, bfr.v,
                                                              (short)0, acc[ti], false, false);
        }
        __syncthreads();
    }

    // Epilogue: bias, q-scale, scatter to q[bh][n][d], k[bh][n][d], vT[bh][d][m]
    #pragma unroll
    for (int ti = 0; ti < 4; ++ti) {
        int c     = n0 + wn * 64 + ti * 16 + (lane & 15);  // 0..2303
        int which = c / 768;
        int cc    = c % 768;
        float bias = (which == 0) ? qb[cc] : (which == 2 ? vb[cc] : 0.0f);
        int hh = cc >> 6, d = cc & 63;
        #pragma unroll
        for (int j = 0; j < 8; ++j) {
            int m = m0 + wm * 16 + j + ((lane >> 4) << 3);  // < 12608
            int b = m / 197, n = m % 197;
            int bh = b * 12 + hh;
            float val = acc[ti][j] + bias;
            if (which == 0)      qbuf[((size_t)bh * 208 + n) * 64 + d] = f2bf(val * 0.125f);
            else if (which == 1) kbuf[((size_t)bh * 224 + n) * 64 + d] = f2bf(val);
            else                 vbuf[((size_t)bh * 64 + d) * 224 + n] = f2bf(val);
        }
    }
}

// ---------------------------------------------------------------------------
// Flash attention: 1 wave per (b,h, 16-row tile). Keys looped 224 in 32s.
// ---------------------------------------------------------------------------
__global__ __launch_bounds__(32)
void attn_kernel(const bf16* __restrict__ qbuf, const bf16* __restrict__ kbuf,
                 const bf16* __restrict__ vbuf, const float* __restrict__ rpb,
                 bf16* __restrict__ obuf)
{
    __shared__ bf16 p_lds[16 * 32];
    const int lane = threadIdx.x;
    const int idx  = blockIdx.x;
    const int tile = idx % 13;
    const int bh   = idx / 13;
    const int b    = bh / 12, h = bh % 12;
    const int n0   = tile * 16;

    const int abase    = (lane < 16) ? 0 : 16;
    const int bbase    = (lane < 16) ? 0 : 16;
    const int myrow_hi = (lane >> 4) << 3;

    const bf16* qp = qbuf + ((size_t)bh * 208 + n0 + (lane & 15)) * 64;
    FragU q0, q1;
    q0.h[0] = *(const v8bf*)(qp + abase);
    q0.h[1] = *(const v8bf*)(qp + abase + 8);
    q1.h[0] = *(const v8bf*)(qp + 32 + abase);
    q1.h[1] = *(const v8bf*)(qp + 32 + abase + 8);

    float mrow[8], lrow[8];
    #pragma unroll
    for (int j = 0; j < 8; ++j) { mrow[j] = -3.0e38f; lrow[j] = 0.f; }
    v8f o0 = {}, o1 = {}, o2 = {}, o3 = {};

    const float* rp = rpb + (size_t)h * 208 * 224;

    for (int mb = 0; mb < 224; mb += 32) {
        v8f s0 = {}, s1 = {};
        {
            const bf16* kp = kbuf + ((size_t)bh * 224 + mb + (lane & 15)) * 64;
            FragU kA, kB;
            kA.h[0] = *(const v8bf*)(kp + bbase);
            kA.h[1] = *(const v8bf*)(kp + bbase + 8);
            kB.h[0] = *(const v8bf*)(kp + 32 + bbase);
            kB.h[1] = *(const v8bf*)(kp + 32 + bbase + 8);
            s0 = __builtin_amdgcn_wmma_f32_16x16x32_bf16(false, q0.v, false, kA.v,
                                                         (short)0, s0, false, false);
            s0 = __builtin_amdgcn_wmma_f32_16x16x32_bf16(false, q1.v, false, kB.v,
                                                         (short)0, s0, false, false);
        }
        {
            const bf16* kp = kbuf + ((size_t)bh * 224 + mb + 16 + (lane & 15)) * 64;
            FragU kA, kB;
            kA.h[0] = *(const v8bf*)(kp + bbase);
            kA.h[1] = *(const v8bf*)(kp + bbase + 8);
            kB.h[0] = *(const v8bf*)(kp + 32 + bbase);
            kB.h[1] = *(const v8bf*)(kp + 32 + bbase + 8);
            s1 = __builtin_amdgcn_wmma_f32_16x16x32_bf16(false, q0.v, false, kA.v,
                                                         (short)0, s1, false, false);
            s1 = __builtin_amdgcn_wmma_f32_16x16x32_bf16(false, q1.v, false, kB.v,
                                                         (short)0, s1, false, false);
        }

        // bias + mask via precomputed padded table (coalesced, branch-free)
        #pragma unroll
        for (int j = 0; j < 8; ++j) {
            const float* rowp = rp + (size_t)(n0 + j + myrow_hi) * 224 + mb;
            s0[j] += rowp[lane & 15];
            s1[j] += rowp[16 + (lane & 15)];
        }

        // online softmax (row stats replicated in 16-lane halves)
        #pragma unroll
        for (int j = 0; j < 8; ++j) {
            float mx = fmaxf(s0[j], s1[j]);
            mx = fmaxf(mx, __shfl_xor(mx, 1, 32));
            mx = fmaxf(mx, __shfl_xor(mx, 2, 32));
            mx = fmaxf(mx, __shfl_xor(mx, 4, 32));
            mx = fmaxf(mx, __shfl_xor(mx, 8, 32));
            float mnew  = fmaxf(mrow[j], mx);
            float scale = __expf(mrow[j] - mnew);
            float p0 = __expf(s0[j] - mnew);
            float p1 = __expf(s1[j] - mnew);
            float rs = p0 + p1;
            rs += __shfl_xor(rs, 1, 32);
            rs += __shfl_xor(rs, 2, 32);
            rs += __shfl_xor(rs, 4, 32);
            rs += __shfl_xor(rs, 8, 32);
            lrow[j] = lrow[j] * scale + rs;
            mrow[j] = mnew;
            o0[j] *= scale; o1[j] *= scale; o2[j] *= scale; o3[j] *= scale;
            s0[j] = p0; s1[j] = p1;
        }

        // repack P (C layout) -> A fragment via LDS
        __syncthreads();
        #pragma unroll
        for (int j = 0; j < 8; ++j) {
            int rr = j + myrow_hi;
            p_lds[rr * 32 + (lane & 15)]      = f2bf(s0[j]);
            p_lds[rr * 32 + 16 + (lane & 15)] = f2bf(s1[j]);
        }
        __syncthreads();
        FragU pf;
        pf.h[0] = *(const v8bf*)(p_lds + (lane & 15) * 32 + abase);
        pf.h[1] = *(const v8bf*)(p_lds + (lane & 15) * 32 + abase + 8);

        // O += P V (vT layout -> contiguous B fragments)
        #pragma unroll
        for (int t2 = 0; t2 < 4; ++t2) {
            const bf16* vp = vbuf + ((size_t)bh * 64 + t2 * 16 + (lane & 15)) * 224 + mb + bbase;
            FragU vf; vf.h[0] = *(const v8bf*)(vp); vf.h[1] = *(const v8bf*)(vp + 8);
            v8f* op = (t2 == 0) ? &o0 : (t2 == 1) ? &o1 : (t2 == 2) ? &o2 : &o3;
            *op = __builtin_amdgcn_wmma_f32_16x16x32_bf16(false, pf.v, false, vf.v,
                                                          (short)0, *op, false, false);
        }
    }

    #pragma unroll
    for (int j = 0; j < 8; ++j) {
        int n = n0 + j + myrow_hi;
        if (n >= 197) continue;
        float inv = 1.0f / lrow[j];
        bf16* op = obuf + ((size_t)(b * 197 + n)) * 768 + h * 64 + (lane & 15);
        op[0]  = f2bf(o0[j] * inv);
        op[16] = f2bf(o1[j] * inv);
        op[32] = f2bf(o2[j] * inv);
        op[48] = f2bf(o3[j] * inv);
    }
}

// ---------------------------------------------------------------------------
// Proj GEMM: o[12608 x 768] bf16 x proj_w^T + proj_b -> out f32.
// Block = 64(M) x 128(N).  A tile copied via global_load_async_to_lds_b128.
// ---------------------------------------------------------------------------
__global__ __launch_bounds__(256)
void proj_gemm(const bf16* __restrict__ a, const float* __restrict__ w,
               const float* __restrict__ pb, float* __restrict__ out)
{
    __shared__ bf16 a_lds[64 * 32];
    __shared__ bf16 b_lds[128 * 32];
    const int t    = threadIdx.x;
    const int lane = t & 31;
    const int wave = t >> 5;
    const int m0   = (int)blockIdx.x * 64;
    const int n0   = (int)blockIdx.y * 128;
    const int wm   = wave >> 1;
    const int wn   = wave & 1;

    v8f acc[4] = {};

    const int ra = t >> 2, ka = (t & 3) * 8;    // A: 16B/thread (async)
    const int rb = t >> 1, kb = (t & 1) * 16;   // B: 16 f32/thread
    const int abase = (lane < 16) ? 0 : 16;
    const int bbase = (lane < 16) ? 0 : 16;
    const int arow  = wm * 16 + (lane & 15);

    const unsigned lds_a = (unsigned)(uintptr_t)(&a_lds[0]) + (unsigned)(t * 16);

    for (int k0 = 0; k0 < 768; k0 += 32) {
        {   // A tile: async bf16 copy global -> LDS, tracked by ASYNCcnt
            unsigned goff = (unsigned)(((m0 + ra) * 768 + k0 + ka) * 2);
            asm volatile("global_load_async_to_lds_b128 %0, %1, %2"
                         :: "v"(lds_a), "v"(goff), "s"(a) : "memory");
        }
        {   // B tile: proj_w f32 -> bf16 LDS
            const float* sb = w + (size_t)(n0 + rb) * 768 + k0 + kb;
            float4 g0 = *(const float4*)(sb);
            float4 g1 = *(const float4*)(sb + 4);
            float4 g2 = *(const float4*)(sb + 8);
            float4 g3 = *(const float4*)(sb + 12);
            *(v8bf*)(b_lds + rb * 32 + kb)     = cvt8(g0, g1);
            *(v8bf*)(b_lds + rb * 32 + kb + 8) = cvt8(g2, g3);
        }
        asm volatile("s_wait_asynccnt 0x0" ::: "memory");
        __syncthreads();

        FragU af;
        af.h[0] = *(const v8bf*)(a_lds + arow * 32 + abase);
        af.h[1] = *(const v8bf*)(a_lds + arow * 32 + abase + 8);
        #pragma unroll
        for (int ti = 0; ti < 4; ++ti) {
            int bc = wn * 64 + ti * 16 + (lane & 15);
            FragU bfr;
            bfr.h[0] = *(const v8bf*)(b_lds + bc * 32 + bbase);
            bfr.h[1] = *(const v8bf*)(b_lds + bc * 32 + bbase + 8);
            acc[ti] = __builtin_amdgcn_wmma_f32_16x16x32_bf16(false, af.v, false, bfr.v,
                                                              (short)0, acc[ti], false, false);
        }
        __syncthreads();
    }

    #pragma unroll
    for (int ti = 0; ti < 4; ++ti) {
        int c = n0 + wn * 64 + ti * 16 + (lane & 15);
        float bias = pb[c];
        #pragma unroll
        for (int j = 0; j < 8; ++j) {
            int m = m0 + wm * 16 + j + ((lane >> 4) << 3);
            out[(size_t)m * 768 + c] = acc[ti][j] + bias;
        }
    }
}

// ---------------------------------------------------------------------------
extern "C" void kernel_launch(void* const* d_in, const int* in_sizes, int n_in,
                              void* d_out, int out_size, void* d_ws, size_t ws_size,
                              hipStream_t stream) {
    (void)in_sizes; (void)n_in; (void)out_size; (void)ws_size;
    const float* x         = (const float*)d_in[0];
    const float* qkv_w     = (const float*)d_in[1];
    const float* q_bias    = (const float*)d_in[2];
    const float* v_bias    = (const float*)d_in[3];
    const float* rel_table = (const float*)d_in[4];
    const float* proj_w    = (const float*)d_in[5];
    const float* proj_b    = (const float*)d_in[6];
    const int*   rel_index = (const int*)d_in[7];
    float* out = (float*)d_out;

    // ws: q(768*208*64) k(768*224*64) vT(768*64*224) o(12608*768) bf16 + rpb f32
    bf16* qbuf = (bf16*)d_ws;
    bf16* kbuf = qbuf + (size_t)768 * 208 * 64;
    bf16* vbuf = kbuf + (size_t)768 * 224 * 64;
    bf16* obuf = vbuf + (size_t)768 * 224 * 64;
    float* rpb = (float*)(obuf + (size_t)12608 * 768);   // 12*208*224 f32

    zero_pads<<<(768 * 27 * 64 + 255) / 256, 256, 0, stream>>>(qbuf, kbuf, vbuf);
    rpb_kernel<<<(12 * 208 * 224 + 255) / 256, 256, 0, stream>>>(rel_table, rel_index, rpb);

    dim3 g1(197, 18);   // 12608/64 x 2304/128
    qkv_gemm<<<g1, 256, 0, stream>>>(x, qkv_w, q_bias, v_bias, qbuf, kbuf, vbuf);

    attn_kernel<<<64 * 12 * 13, 32, 0, stream>>>(qbuf, kbuf, vbuf, rpb, obuf);

    dim3 g3(197, 6);    // 12608/64 x 768/128
    proj_gemm<<<g3, 256, 0, stream>>>(obuf, proj_w, proj_b, out);
}
